// BinderGNN_87067577024730
// MI455X (gfx1250) — matrix operand order repaired and checked by reading, not compiled, not run on hardware
//
#include <hip/hip_runtime.h>
#include <hip/hip_bf16.h>
#include <math.h>

// ---------------------------------------------------------------------------
// BinderGNN forward for MI455X (gfx1250, wave32, WMMA bf16).
// All big GEMMs use V_WMMA_F32_16X16X32_BF16 with fused LDS-resident hidden
// layers to stay under the 23.3 TB/s HBM roof.
// ---------------------------------------------------------------------------

#define Bc    4
#define Nn    1024
#define Hd    128
#define TK    30
#define Ll    3
#define NHh   4
#define NKc   (Nn*TK)        // 30720
#define BNKc  (Bc*NKc)       // 122880
#define BNc   (Bc*Nn)        // 4096

typedef __bf16 bf16_t;
typedef __attribute__((ext_vector_type(8)))  bf16_t v8bf;
typedef __attribute__((ext_vector_type(16))) bf16_t v16bf;
typedef __attribute__((ext_vector_type(8)))  float  v8f;

__device__ __forceinline__ bf16_t f2bf(float f) {
    union { float f; unsigned int u; } c; c.f = f;
    unsigned int u = c.u;
    unsigned int r = (u + 0x7FFFu + ((u >> 16) & 1u)) >> 16;
    union { unsigned short s; bf16_t b; } o; o.s = (unsigned short)r;
    return o.b;
}

__device__ __forceinline__ v16bf mk16(v8bf lo, v8bf hi) {
    return __builtin_shufflevector(lo, hi, 0,1,2,3,4,5,6,7,8,9,10,11,12,13,14,15);
}

__device__ __forceinline__ v8f wmma_bf16(v16bf a, v16bf b, v8f c) {
    return __builtin_amdgcn_wmma_f32_16x16x32_bf16(false, a, false, b, (short)0, c, false, false);
}

// Branch-free GELU (tanh form, single v_exp_f32, no exec-mask divergence).
// tanh(u) = 1 - 2/(exp(2u)+1): exact at both saturations, no NaN path.
__device__ __forceinline__ float gelu_f(float x) {
    float u  = 0.7978845608028654f * (x + 0.044715f * x * x * x);
    float e  = __expf(2.0f * u);
    float th = 1.0f - 2.0f / (e + 1.0f);
    return 0.5f * x * (1.0f + th);
}

// --------------------------------------------------------------------------
// Weight pre-swizzle into wave32 WMMA B-fragment layout:
// lane L holds 16 bf16 = K rows ks*32 + (L>>4)*8 + {0..7, 16..23} of
// column ntile*16 + (L&15).  One 32B contiguous load per fragment.
// --------------------------------------------------------------------------
__global__ void pack_kernel(const float* __restrict__ W, bf16_t* __restrict__ out,
                            int Kd, int Nd) {
    int tid = blockIdx.x * blockDim.x + threadIdx.x;
    int total = Kd * Nd;
    if (tid >= total) return;
    int e    = tid & 15;
    int lane = (tid >> 4) & 31;
    int tile = tid >> 9;
    int ntiles = Nd >> 4;
    int nt = tile % ntiles;
    int ks = tile / ntiles;
    int k   = ks * 32 + ((e >> 3) << 4) + ((lane >> 4) << 3) + (e & 7);
    int col = nt * 16 + (lane & 15);
    out[tid] = f2bf(W[(size_t)k * Nd + col]);
}

// --------------------------------------------------------------------------
// KNN (top-30 smallest masked distances, stable like jax top_k)
// --------------------------------------------------------------------------
__global__ void knn_kernel(const float* __restrict__ X, const float* __restrict__ mask,
                           float* __restrict__ E, int* __restrict__ Eidx) {
    int tid = blockIdx.x * blockDim.x + threadIdx.x;
    if (tid >= Bc * Nn) return;
    int b = tid >> 10, i = tid & 1023;
    const float* Xb = X + (size_t)b * Nn * 3;
    float xi0 = Xb[i*3], xi1 = Xb[i*3+1], xi2 = Xb[i*3+2];
    float mi = mask[b*Nn + i];
    float dmax = 0.f;
    for (int j = 0; j < Nn; ++j) {
        float dx = xi0 - Xb[j*3], dy = xi1 - Xb[j*3+1], dz = xi2 - Xb[j*3+2];
        float d  = sqrtf(dx*dx + dy*dy + dz*dz + 1e-6f);
        float m2 = mi * mask[b*Nn + j];
        dmax = fmaxf(dmax, m2 * d);
    }
    float val[TK]; int idx[TK]; int cnt = 0;
    for (int j = 0; j < Nn; ++j) {
        float dx = xi0 - Xb[j*3], dy = xi1 - Xb[j*3+1], dz = xi2 - Xb[j*3+2];
        float d  = sqrtf(dx*dx + dy*dy + dz*dz + 1e-6f);
        float m2 = mi * mask[b*Nn + j];
        float Da = m2 * d + (1.f - m2) * dmax;
        if (cnt < TK || Da < val[TK-1]) {
            int pos = (cnt < TK) ? cnt : TK - 1;
            while (pos > 0 && val[pos-1] > Da) { val[pos] = val[pos-1]; idx[pos] = idx[pos-1]; --pos; }
            val[pos] = Da; idx[pos] = j;
            if (cnt < TK) ++cnt;
        }
    }
    for (int k = 0; k < TK; ++k) {
        E[(size_t)tid*TK + k]    = val[k];
        Eidx[(size_t)tid*TK + k] = idx[k];
    }
}

// -------------------- edge-dedup: rank over value space ---------------------
__global__ void zero_flags_kernel(int* __restrict__ f, int n) {
    for (int i = blockIdx.x * blockDim.x + threadIdx.x; i < n; i += gridDim.x * blockDim.x) f[i] = 0;
}

__global__ void ei_flags_kernel(const int* __restrict__ Eidx, int* __restrict__ ei,
                                int* __restrict__ flags) {
    int m = blockIdx.x * blockDim.x + threadIdx.x;
    if (m >= BNKc) return;
    int b = m / NKc, r = m - b * NKc, n = r / TK;
    int j = Eidx[m];
    int lo = j < n ? j : n, hi = j < n ? n : j;
    int v = lo * Nn + hi;
    ei[m] = v;
    flags[(b << 20) + v] = 1;
}

__global__ void scan_kernel(int* __restrict__ flags) {
    __shared__ int sh[1024];
    int b = blockIdx.x, t = threadIdx.x;
    int carry = 0;
    for (int c = 0; c < 1024; ++c) {
        int pos = (b << 20) + c * 1024 + t;
        int v = flags[pos];
        sh[t] = v; __syncthreads();
        int run = v;
        for (int o = 1; o < 1024; o <<= 1) {
            int x = (t >= o) ? sh[t - o] : 0;
            __syncthreads();
            run += x; sh[t] = run;
            __syncthreads();
        }
        flags[pos] = carry + run - v;     // exclusive rank
        carry += sh[1023];
        __syncthreads();
    }
}

__global__ void inv_kernel(const int* __restrict__ ei, const int* __restrict__ rank,
                           int* __restrict__ inv) {
    int m = blockIdx.x * blockDim.x + threadIdx.x;
    if (m >= BNKc) return;
    int b = m / NKc;
    inv[m] = rank[(b << 20) + ei[m]];
}

__global__ void maskat_kernel(const int* __restrict__ Eidx, const float* __restrict__ mask,
                              float* __restrict__ ma) {
    int m = blockIdx.x * blockDim.x + threadIdx.x;
    if (m >= BNKc) return;
    int b = m / NKc, r = m - b * NKc, n = r / TK;
    ma[m] = mask[b*Nn + n] * mask[b*Nn + Eidx[m]];
}

// -------------------- misc element-wise kernels ----------------------------
__global__ void hEinit_kernel(const float* __restrict__ E, const float* __restrict__ Ww,
                              const float* __restrict__ Wb, float* __restrict__ hE) {
    int tid = blockIdx.x * blockDim.x + threadIdx.x;
    int row = tid >> 7, h = tid & 127;
    hE[tid] = gelu_f(E[row] * Ww[h] + Wb[h]);
}

__global__ void merge_kernel(const float* __restrict__ src, const int* __restrict__ inv,
                             float* __restrict__ dstf, bf16_t* __restrict__ dstb) {
    int tid = blockIdx.x * blockDim.x + threadIdx.x;
    int bm = tid >> 7, h = tid & 127;
    int b = bm / NKc, m = bm - b * NKc;
    float v = src[((size_t)(b * NKc + inv[bm])) * 128 + h];
    dstf[tid] = v;
    dstb[tid] = f2bf(v);
}

__global__ void vinit_kernel(const int* __restrict__ rla, const float* __restrict__ embed,
                             float* __restrict__ Vf, bf16_t* __restrict__ Vb) {
    int tid = blockIdx.x * blockDim.x + threadIdx.x;
    int node = tid >> 7, h = tid & 127;
    float v = embed[(size_t)rla[node] * 128 + h];
    Vf[tid] = v; Vb[tid] = f2bf(v);
}

__global__ void rlamean_kernel(const int* __restrict__ rla, const float* __restrict__ mask,
                               const float* __restrict__ embed, float* __restrict__ hT) {
    __shared__ float rs[128], ms[128];
    __shared__ int mean_s;
    int b = blockIdx.x, t = threadIdx.x;
    float sr = 0.f, sm = 0.f;
    for (int n = t; n < Nn; n += 128) { sr += (float)rla[b*Nn + n]; sm += mask[b*Nn + n]; }
    rs[t] = sr; ms[t] = sm; __syncthreads();
    for (int o = 64; o > 0; o >>= 1) { if (t < o) { rs[t] += rs[t+o]; ms[t] += ms[t+o]; } __syncthreads(); }
    if (t == 0) mean_s = (int)truncf(rs[0] / ms[0]);
    __syncthreads();
    hT[b*128 + t] = embed[(size_t)mean_s * 128 + t];
}

// --------------------------------------------------------------------------
// Fused 3-layer MLP (384 -> relu 128 -> relu 128 -> 128), gather-on-the-fly.
// MODE 0: edge (V[e0] | V[ej] | h_E)    MODE 1: node (V[n] | h_E | V[ej])
// One wave per 16-row tile; hidden tiles live in per-wave LDS.
// --------------------------------------------------------------------------
template<int MODE>
__global__ void __launch_bounds__(256) mlp3_kernel(
        const bf16_t* __restrict__ Vb, const bf16_t* __restrict__ hEb,
        const int* __restrict__ Eidx,
        const bf16_t* __restrict__ W1p, const float* __restrict__ b1,
        const bf16_t* __restrict__ W2p, const float* __restrict__ b2,
        const bf16_t* __restrict__ W3p, const float* __restrict__ b3,
        const float* __restrict__ maskat, float* __restrict__ out) {
    __shared__ __align__(16) bf16_t hid[8][16][128];
    int lane = threadIdx.x & 31, w = threadIdx.x >> 5;
    int tile = blockIdx.x * 8 + w;
    int row0 = tile * 16;
    int hf = lane >> 4, cl = lane & 15;
    int row = row0 + cl;
    int b = row / NKc, rem = row - b * NKc, n = rem / TK;
    int e0 = Eidx[(b*Nn + n)*TK];
    int ej = Eidx[row];
    const bf16_t *p0, *p1, *p2;
    if (MODE == 0) {
        p0 = Vb  + ((size_t)b*Nn + e0) * Hd;
        p1 = Vb  + ((size_t)b*Nn + ej) * Hd;
        p2 = hEb + (size_t)row * Hd;
    } else {
        p0 = Vb  + ((size_t)b*Nn + n) * Hd;
        p1 = hEb + (size_t)row * Hd;
        p2 = Vb  + ((size_t)b*Nn + ej) * Hd;
    }
    auto ch = [&](int ka) -> v8bf {
        const bf16_t* p = (ka < 128) ? p0 + ka : (ka < 256) ? p1 + (ka - 128) : p2 + (ka - 256);
        return *(const v8bf*)p;
    };
    const v8f vz = {0,0,0,0,0,0,0,0};
    v8f acc[8];
    // ---- stage 1: K=384 ----
    #pragma unroll
    for (int nt = 0; nt < 8; ++nt) acc[nt] = vz;
    for (int ks = 0; ks < 12; ++ks) {
        int base = ks * 32 + hf * 8;
        v16bf a = mk16(ch(base), ch(base + 16));
        #pragma unroll
        for (int nt = 0; nt < 8; ++nt) {
            v16bf bf = *(const v16bf*)(W1p + (((size_t)ks*8 + nt)*32 + lane)*16);
            acc[nt] = wmma_bf16(a, bf, acc[nt]);
        }
    }
    #pragma unroll
    for (int nt = 0; nt < 8; ++nt)
        #pragma unroll
        for (int r = 0; r < 8; ++r) {
            int col = nt*16 + cl, rl = r + hf*8;
            float v = acc[nt][r] + b1[col];
            hid[w][rl][col] = f2bf(v > 0.f ? v : 0.f);
        }
    __syncthreads();
    // ---- stage 2: K=128 ----
    #pragma unroll
    for (int nt = 0; nt < 8; ++nt) acc[nt] = vz;
    for (int ks = 0; ks < 4; ++ks) {
        const bf16_t* pr = &hid[w][cl][ks*32 + hf*8];
        v16bf a = mk16(*(const v8bf*)pr, *(const v8bf*)(pr + 16));
        #pragma unroll
        for (int nt = 0; nt < 8; ++nt) {
            v16bf bf = *(const v16bf*)(W2p + (((size_t)ks*8 + nt)*32 + lane)*16);
            acc[nt] = wmma_bf16(a, bf, acc[nt]);
        }
    }
    __syncthreads();
    #pragma unroll
    for (int nt = 0; nt < 8; ++nt)
        #pragma unroll
        for (int r = 0; r < 8; ++r) {
            int col = nt*16 + cl, rl = r + hf*8;
            float v = acc[nt][r] + b2[col];
            hid[w][rl][col] = f2bf(v > 0.f ? v : 0.f);
        }
    __syncthreads();
    // ---- stage 3: K=128, masked output ----
    #pragma unroll
    for (int nt = 0; nt < 8; ++nt) acc[nt] = vz;
    for (int ks = 0; ks < 4; ++ks) {
        const bf16_t* pr = &hid[w][cl][ks*32 + hf*8];
        v16bf a = mk16(*(const v8bf*)pr, *(const v8bf*)(pr + 16));
        #pragma unroll
        for (int nt = 0; nt < 8; ++nt) {
            v16bf bf = *(const v16bf*)(W3p + (((size_t)ks*8 + nt)*32 + lane)*16);
            acc[nt] = wmma_bf16(a, bf, acc[nt]);
        }
    }
    float mrow[8];
    #pragma unroll
    for (int r = 0; r < 8; ++r) mrow[r] = maskat[row0 + r + hf*8];
    #pragma unroll
    for (int nt = 0; nt < 8; ++nt)
        #pragma unroll
        for (int r = 0; r < 8; ++r) {
            int grow = row0 + r + hf*8;
            int col = nt*16 + cl;
            out[(size_t)grow*128 + col] = (acc[nt][r] + b3[col]) * mrow[r];
        }
}

// --------------------------------------------------------------------------
// Fused FFN (128 -> gelu 512 -> 128), hidden tile in LDS. 2 waves / block.
// --------------------------------------------------------------------------
__global__ void __launch_bounds__(64) ffn_kernel(
        const bf16_t* __restrict__ X,
        const bf16_t* __restrict__ Fip, const float* __restrict__ bi,
        const bf16_t* __restrict__ Fop, const float* __restrict__ bo,
        float* __restrict__ out) {
    __shared__ __align__(16) bf16_t hid[2][16][512];
    int lane = threadIdx.x & 31, w = threadIdx.x >> 5;
    int tile = blockIdx.x * 2 + w;
    int row0 = tile * 16;
    int hf = lane >> 4, cl = lane & 15;
    v16bf a[4];
    #pragma unroll
    for (int ks = 0; ks < 4; ++ks) {
        const bf16_t* pr = X + (size_t)(row0 + cl)*128 + ks*32 + hf*8;
        a[ks] = mk16(*(const v8bf*)pr, *(const v8bf*)(pr + 16));
    }
    const v8f vz = {0,0,0,0,0,0,0,0};
    for (int nt = 0; nt < 32; ++nt) {
        // keep the next tile's packed weights moving toward L0
        if (nt + 1 < 32)
            __builtin_prefetch(Fip + (((size_t)(nt + 1))*32 + lane)*16, 0, 1);
        v8f c = vz;
        #pragma unroll
        for (int ks = 0; ks < 4; ++ks) {
            v16bf bf = *(const v16bf*)(Fip + (((size_t)ks*32 + nt)*32 + lane)*16);
            c = wmma_bf16(a[ks], bf, c);
        }
        #pragma unroll
        for (int r = 0; r < 8; ++r) {
            int col = nt*16 + cl, rl = r + hf*8;
            hid[w][rl][col] = f2bf(gelu_f(c[r] + bi[col]));
        }
    }
    __syncthreads();
    v8f acc[8];
    #pragma unroll
    for (int nt = 0; nt < 8; ++nt) acc[nt] = vz;
    for (int ks = 0; ks < 16; ++ks) {
        if (ks + 1 < 16)
            __builtin_prefetch(Fop + (((size_t)(ks + 1)*8)*32 + lane)*16, 0, 1);
        const bf16_t* pr = &hid[w][cl][ks*32 + hf*8];
        v16bf af = mk16(*(const v8bf*)pr, *(const v8bf*)(pr + 16));
        #pragma unroll
        for (int nt = 0; nt < 8; ++nt) {
            v16bf bf = *(const v16bf*)(Fop + (((size_t)ks*8 + nt)*32 + lane)*16);
            acc[nt] = wmma_bf16(af, bf, acc[nt]);
        }
    }
    #pragma unroll
    for (int nt = 0; nt < 8; ++nt)
        #pragma unroll
        for (int r = 0; r < 8; ++r) {
            int grow = row0 + r + hf*8;
            int col = nt*16 + cl;
            out[(size_t)grow*128 + col] = acc[nt][r] + bo[col];
        }
}

// -------------------- generic M x K x 128 WMMA GEMM ------------------------
__global__ void __launch_bounds__(256) gemm128_kernel(
        const bf16_t* __restrict__ A, int Kd,
        const bf16_t* __restrict__ Bp, const float* __restrict__ bias,
        float* __restrict__ C) {
    int lane = threadIdx.x & 31, w = threadIdx.x >> 5;
    int tile = blockIdx.x * 8 + w;
    int row0 = tile * 16;
    int hf = lane >> 4, cl = lane & 15;
    const v8f vz = {0,0,0,0,0,0,0,0};
    v8f acc[8];
    #pragma unroll
    for (int nt = 0; nt < 8; ++nt) acc[nt] = vz;
    int ksteps = Kd >> 5;
    for (int ks = 0; ks < ksteps; ++ks) {
        if (ks + 1 < ksteps)
            __builtin_prefetch(Bp + (((size_t)(ks + 1)*8)*32 + lane)*16, 0, 1);
        const bf16_t* pr = A + (size_t)(row0 + cl)*Kd + ks*32 + hf*8;
        v16bf a = mk16(*(const v8bf*)pr, *(const v8bf*)(pr + 16));
        #pragma unroll
        for (int nt = 0; nt < 8; ++nt) {
            v16bf bf = *(const v16bf*)(Bp + (((size_t)ks*8 + nt)*32 + lane)*16);
            acc[nt] = wmma_bf16(a, bf, acc[nt]);
        }
    }
    #pragma unroll
    for (int nt = 0; nt < 8; ++nt)
        #pragma unroll
        for (int r = 0; r < 8; ++r) {
            int grow = row0 + r + hf*8;
            int col = nt*16 + cl;
            float v = acc[nt][r] + (bias ? bias[col] : 0.f);
            C[(size_t)grow*128 + col] = v;
        }
}

// -------------------- LayerNorm (row width 128) ----------------------------
__global__ void ln_kernel(const float* __restrict__ base, const float* __restrict__ add,
                          const float* __restrict__ s, const float* __restrict__ bv,
                          const float* __restrict__ rowmask, int mdiv,
                          float* __restrict__ outf, bf16_t* __restrict__ outb) {
    __shared__ float red[128];
    int row = blockIdx.x, t = threadIdx.x;
    float x = base[(size_t)row*128 + t] + add[(size_t)row*128 + t];
    red[t] = x; __syncthreads();
    for (int o = 64; o > 0; o >>= 1) { if (t < o) red[t] += red[t+o]; __syncthreads(); }
    float mu = red[0] * (1.f/128.f); __syncthreads();
    float d = x - mu;
    red[t] = d * d; __syncthreads();
    for (int o = 64; o > 0; o >>= 1) { if (t < o) red[t] += red[t+o]; __syncthreads(); }
    float var = red[0] * (1.f/128.f);
    float y = d * rsqrtf(var + 1e-5f) * s[t] + bv[t];
    if (rowmask) y *= rowmask[row / mdiv];
    outf[(size_t)row*128 + t] = y;
    if (outb) outb[(size_t)row*128 + t] = f2bf(y);
}

// node message reduce over k (/=SCALE) fused with LN1
__global__ void node_reduce_ln_kernel(const float* __restrict__ m3, const float* __restrict__ Vin,
                                      const float* __restrict__ s, const float* __restrict__ bv,
                                      float* __restrict__ Vf, bf16_t* __restrict__ Vb) {
    __shared__ float red[128];
    int node = blockIdx.x, t = threadIdx.x;
    float sum = 0.f;
    for (int kk = 0; kk < TK; ++kk) sum += m3[((size_t)node*TK + kk)*128 + t];
    float x = Vin[(size_t)node*128 + t] + sum * (1.f/30.0f);
    red[t] = x; __syncthreads();
    for (int o = 64; o > 0; o >>= 1) { if (t < o) red[t] += red[t+o]; __syncthreads(); }
    float mu = red[0] * (1.f/128.f); __syncthreads();
    float d = x - mu;
    red[t] = d * d; __syncthreads();
    for (int o = 64; o > 0; o >>= 1) { if (t < o) red[t] += red[t+o]; __syncthreads(); }
    float var = red[0] * (1.f/128.f);
    float y = d * rsqrtf(var + 1e-5f) * s[t] + bv[t];
    Vf[(size_t)node*128 + t] = y;
    Vb[(size_t)node*128 + t] = f2bf(y);
}

// -------------------- token attention path ---------------------------------
__global__ void hin_kernel(const float* __restrict__ hT, const float* __restrict__ Vf,
                           bf16_t* __restrict__ hin) {
    int tid = blockIdx.x * blockDim.x + threadIdx.x;
    int node = tid >> 8, c = tid & 255;
    int b = node >> 10;
    float v = (c < 128) ? hT[b*128 + c] : Vf[(size_t)node*128 + (c - 128)];
    hin[tid] = f2bf(v);
}

__global__ void q_kernel(const float* __restrict__ hT, const float* __restrict__ WQ,
                         float* __restrict__ Q) {
    int b = blockIdx.x, o = threadIdx.x;
    float a = 0.f;
    for (int i = 0; i < 128; ++i) a += hT[b*128 + i] * WQ[i*128 + o];
    Q[b*128 + o] = a;
}

__global__ void attn_kernel(const float* __restrict__ Q, const float* __restrict__ Kt,
                            const float* __restrict__ Vt, const float* __restrict__ mask,
                            float* __restrict__ upd) {
    __shared__ float sm[1024];
    __shared__ float red[256];
    int b = blockIdx.x >> 2, hh = blockIdx.x & 3;
    int t = threadIdx.x;
    float q[32];
    #pragma unroll
    for (int d = 0; d < 32; ++d) q[d] = Q[b*128 + hh*32 + d];
    float lmax = -3.402823466e38f;
    for (int n = t; n < 1024; n += 256) {
        const float* kp = Kt + ((size_t)(b*Nn + n))*128 + hh*32;
        float dot = 0.f;
        #pragma unroll
        for (int d = 0; d < 32; ++d) dot += q[d] * kp[d];
        float l = dot * 0.17677669529663687f;   // 1/sqrt(32)
        float mv = mask[b*Nn + n];
        l = (mv > 0.f) ? l : -3.402823466e38f;
        sm[n] = l;
        lmax = fmaxf(lmax, l);
    }
    red[t] = lmax; __syncthreads();
    for (int o = 128; o > 0; o >>= 1) { if (t < o) red[t] = fmaxf(red[t], red[t+o]); __syncthreads(); }
    float gmax = red[0]; __syncthreads();
    float lsum = 0.f;
    for (int n = t; n < 1024; n += 256) { float e = __expf(sm[n] - gmax); sm[n] = e; lsum += e; }
    red[t] = lsum; __syncthreads();
    for (int o = 128; o > 0; o >>= 1) { if (t < o) red[t] += red[t+o]; __syncthreads(); }
    float inv = 1.f / red[0]; __syncthreads();
    for (int n = t; n < 1024; n += 256) sm[n] = sm[n] * inv * mask[b*Nn + n];
    __syncthreads();
    int d = t & 31, part = t >> 5;
    float acc = 0.f;
    for (int n = part*128; n < part*128 + 128; ++n)
        acc += sm[n] * Vt[((size_t)(b*Nn + n))*128 + hh*32 + d];
    red[t] = acc; __syncthreads();
    if (part == 0) {
        float s = 0.f;
        #pragma unroll
        for (int p = 0; p < 8; ++p) s += red[p*32 + d];
        upd[b*128 + hh*32 + d] = s;
    }
}

__global__ void token_update_kernel(const float* __restrict__ upd, const float* __restrict__ hTin,
        const float* __restrict__ WO, const float* __restrict__ WOb,
        const float* __restrict__ s1, const float* __restrict__ b1,
        const float* __restrict__ Fi, const float* __restrict__ Fib,
        const float* __restrict__ Fo, const float* __restrict__ Fob,
        const float* __restrict__ s2, const float* __restrict__ b2v,
        int layer, float* __restrict__ hTout) {
    __shared__ float ht[128];
    __shared__ float hid[512];
    __shared__ float red[128];
    int b = blockIdx.x, t = threadIdx.x;
    float x = 0.f;
    if (t < 128) {
        float dh = 0.f;
        for (int i = 0; i < 128; ++i) dh += upd[b*128 + i] * WO[i*128 + t];
        dh += WOb[t];
        x = ((layer > 0) ? hTin[b*128 + t] : 0.f) + dh;
        red[t] = x;
    }
    __syncthreads();
    for (int o = 64; o > 0; o >>= 1) { if (t < o) red[t] += red[t+o]; __syncthreads(); }
    float mu = red[0] * (1.f/128.f); __syncthreads();
    if (t < 128) { float d = x - mu; red[t] = d * d; } __syncthreads();
    for (int o = 64; o > 0; o >>= 1) { if (t < o) red[t] += red[t+o]; __syncthreads(); }
    float var = red[0] * (1.f/128.f); __syncthreads();
    if (t < 128) ht[t] = (x - mu) * rsqrtf(var + 1e-5f) * s1[t] + b1[t];
    __syncthreads();
    { // FFN hidden, t covers all 512
        float a = 0.f;
        for (int i = 0; i < 128; ++i) a += ht[i] * Fi[i*512 + t];
        hid[t] = gelu_f(a + Fib[t]);
    }
    __syncthreads();
    float x2 = 0.f;
    if (t < 128) {
        float v = 0.f;
        for (int j = 0; j < 512; ++j) v += hid[j] * Fo[j*128 + t];
        x2 = ht[t] + v + Fob[t];
        red[t] = x2;
    }
    __syncthreads();
    for (int o = 64; o > 0; o >>= 1) { if (t < o) red[t] += red[t+o]; __syncthreads(); }
    float mu2 = red[0] * (1.f/128.f); __syncthreads();
    if (t < 128) { float d = x2 - mu2; red[t] = d * d; } __syncthreads();
    for (int o = 64; o > 0; o >>= 1) { if (t < o) red[t] += red[t+o]; __syncthreads(); }
    float var2 = red[0] * (1.f/128.f);
    if (t < 128)
        hTout[b*128 + t] = (x2 - mu2) * rsqrtf(var2 + 1e-5f) * s2[t] + b2v[t];
}

__global__ void final_kernel(const float* __restrict__ hT, const float* __restrict__ Ww,
                             const float* __restrict__ Wb, float* __restrict__ out) {
    __shared__ float red[128];
    int t = threadIdx.x;
    for (int b = 0; b < Bc; ++b) {
        red[t] = hT[b*128 + t] * Ww[t]; __syncthreads();
        for (int o = 64; o > 0; o >>= 1) { if (t < o) red[t] += red[t+o]; __syncthreads(); }
        if (t == 0) { float z = red[0] + Wb[0]; out[b] = 1.f / (1.f + __expf(-z)); }
        __syncthreads();
    }
}

// ===========================================================================
extern "C" void kernel_launch(void* const* d_in, const int* in_sizes, int n_in,
                              void* d_out, int out_size, void* d_ws, size_t ws_size,
                              hipStream_t stream) {
    // ---- inputs (setup_inputs dict order) ----
    const int*   rla   = (const int*)  d_in[0];
    const float* X     = (const float*)d_in[1];
    const float* mask  = (const float*)d_in[2];
    const float* embed = (const float*)d_in[3];
    const float* WE_w  = (const float*)d_in[4];
    const float* WE_b  = (const float*)d_in[5];
    const float* nW1 = (const float*)d_in[6],  *nW1b = (const float*)d_in[7];
    const float* nW2 = (const float*)d_in[8],  *nW2b = (const float*)d_in[9];
    const float* nW3 = (const float*)d_in[10], *nW3b = (const float*)d_in[11];
    const float* nLN1s=(const float*)d_in[12], *nLN1b=(const float*)d_in[13];
    const float* nLN2s=(const float*)d_in[14], *nLN2b=(const float*)d_in[15];
    const float* nFi = (const float*)d_in[16], *nFib = (const float*)d_in[17];
    const float* nFo = (const float*)d_in[18], *nFob = (const float*)d_in[19];
    const float* eW1 = (const float*)d_in[20], *eW1b = (const float*)d_in[21];
    const float* eW2 = (const float*)d_in[22], *eW2b = (const float*)d_in[23];
    const float* eW3 = (const float*)d_in[24], *eW3b = (const float*)d_in[25];
    const float* eLN1s=(const float*)d_in[26], *eLN1b=(const float*)d_in[27];
    const float* eLN2s=(const float*)d_in[28], *eLN2b=(const float*)d_in[29];
    const float* eFi = (const float*)d_in[30], *eFib = (const float*)d_in[31];
    const float* eFo = (const float*)d_in[32], *eFob = (const float*)d_in[33];
    const float* tWQ = (const float*)d_in[34];
    const float* tWK = (const float*)d_in[35];
    const float* tWV = (const float*)d_in[36];
    const float* tWO = (const float*)d_in[37], *tWOb = (const float*)d_in[38];
    const float* tLN1s=(const float*)d_in[39], *tLN1b=(const float*)d_in[40];
    const float* tLN2s=(const float*)d_in[41], *tLN2b=(const float*)d_in[42];
    const float* tFi = (const float*)d_in[43], *tFib = (const float*)d_in[44];
    const float* tFo = (const float*)d_in[45], *tFob = (const float*)d_in[46];
    const float* Woutw=(const float*)d_in[47], *Woutb=(const float*)d_in[48];

    // ---- workspace carve-up ----
    char* ws = (char*)d_ws;
    size_t off = 0;
    auto alloc = [&](size_t bytes) -> char* {
        char* p = ws + off;
        off = (off + bytes + 255) & ~(size_t)255;
        return p;
    };
    float*  E      = (float*) alloc((size_t)BNKc * 4);
    int*    Eidx   = (int*)   alloc((size_t)BNKc * 4);
    int*    ei     = (int*)   alloc((size_t)BNKc * 4);
    int*    inv    = (int*)   alloc((size_t)BNKc * 4);
    float*  maskat = (float*) alloc((size_t)BNKc * 4);
    int*    rank   = (int*)   alloc((size_t)Bc * (1 << 20) * 4);
    float*  hEa    = (float*) alloc((size_t)BNKc * 128 * 4);
    float*  hEb_f  = (float*) alloc((size_t)BNKc * 128 * 4);
    bf16_t* hEbf   = (bf16_t*)alloc((size_t)BNKc * 128 * 2);
    float*  Vf     = (float*) alloc((size_t)BNc * 128 * 4);
    bf16_t* Vbf    = (bf16_t*)alloc((size_t)BNc * 128 * 2);
    float*  m3     = (float*) alloc((size_t)BNKc * 128 * 4);
    float*  hT     = (float*) alloc((size_t)Bc * 128 * 4);
    float*  Qb     = (float*) alloc((size_t)Bc * 128 * 4);
    float*  updb   = (float*) alloc((size_t)Bc * 128 * 4);
    bf16_t* hin    = (bf16_t*)alloc((size_t)BNc * 256 * 2);
    float*  Kt     = (float*) alloc((size_t)BNc * 128 * 4);
    float*  Vt     = (float*) alloc((size_t)BNc * 128 * 4);
    bf16_t* pk     = (bf16_t*)alloc((size_t)3 * 491520 * 2);

    // packed-weight offsets per layer
    const size_t oE1=0, oE2=49152, oE3=65536, oEFi=81920, oEFo=147456,
                 oN1=212992, oN2=262144, oN3=278528, oNFi=294912, oNFo=360448,
                 oTK=425984, oTV=458752, PER=491520;

    // ---- pack all WMMA weights ----
    for (int l = 0; l < Ll; ++l) {
        bf16_t* base = pk + (size_t)l * PER;
        pack_kernel<<<(384*128+255)/256,256,0,stream>>>(eW1 + (size_t)l*384*128, base+oE1, 384,128);
        pack_kernel<<<(128*128+255)/256,256,0,stream>>>(eW2 + (size_t)l*128*128, base+oE2, 128,128);
        pack_kernel<<<(128*128+255)/256,256,0,stream>>>(eW3 + (size_t)l*128*128, base+oE3, 128,128);
        pack_kernel<<<(128*512+255)/256,256,0,stream>>>(eFi + (size_t)l*128*512, base+oEFi,128,512);
        pack_kernel<<<(512*128+255)/256,256,0,stream>>>(eFo + (size_t)l*512*128, base+oEFo,512,128);
        pack_kernel<<<(384*128+255)/256,256,0,stream>>>(nW1 + (size_t)l*384*128, base+oN1, 384,128);
        pack_kernel<<<(128*128+255)/256,256,0,stream>>>(nW2 + (size_t)l*128*128, base+oN2, 128,128);
        pack_kernel<<<(128*128+255)/256,256,0,stream>>>(nW3 + (size_t)l*128*128, base+oN3, 128,128);
        pack_kernel<<<(128*512+255)/256,256,0,stream>>>(nFi + (size_t)l*128*512, base+oNFi,128,512);
        pack_kernel<<<(512*128+255)/256,256,0,stream>>>(nFo + (size_t)l*512*128, base+oNFo,512,128);
        pack_kernel<<<(256*128+255)/256,256,0,stream>>>(tWK + (size_t)l*256*128, base+oTK, 256,128);
        pack_kernel<<<(256*128+255)/256,256,0,stream>>>(tWV + (size_t)l*256*128, base+oTV, 256,128);
    }

    // ---- graph construction ----
    knn_kernel<<<(Bc*Nn)/128,128,0,stream>>>(X, mask, E, Eidx);
    zero_flags_kernel<<<4096,256,0,stream>>>(rank, Bc << 20);
    ei_flags_kernel<<<BNKc/256,256,0,stream>>>(Eidx, ei, rank);
    scan_kernel<<<Bc,1024,0,stream>>>(rank);
    inv_kernel<<<BNKc/256,256,0,stream>>>(ei, rank, inv);
    maskat_kernel<<<BNKc/256,256,0,stream>>>(Eidx, mask, maskat);

    // ---- feature init ----
    hEinit_kernel<<<(size_t)BNKc*128/256,256,0,stream>>>(E, WE_w, WE_b, hEa);
    merge_kernel<<<(size_t)BNKc*128/256,256,0,stream>>>(hEa, inv, hEb_f, hEbf);
    vinit_kernel<<<(size_t)BNc*128/256,256,0,stream>>>(rla, embed, Vf, Vbf);
    rlamean_kernel<<<Bc,128,0,stream>>>(rla, mask, embed, hT);

    float* cur = hEb_f;
    float* alt = hEa;

    for (int l = 0; l < Ll; ++l) {
        bf16_t* base = pk + (size_t)l * PER;
        // -------- edge update --------
        mlp3_kernel<0><<<BNKc/16/8,256,0,stream>>>(Vbf, hEbf, Eidx,
            base+oE1, eW1b + l*128, base+oE2, eW2b + l*128, base+oE3, eW3b + l*128,
            maskat, m3);
        ln_kernel<<<BNKc,128,0,stream>>>(cur, m3, eLN1s + l*128, eLN1b + l*128,
            nullptr, 1, cur, hEbf);
        ffn_kernel<<<BNKc/32,64,0,stream>>>(hEbf, base+oEFi, eFib + l*512,
            base+oEFo, eFob + l*128, m3);
        ln_kernel<<<BNKc,128,0,stream>>>(cur, m3, eLN2s + l*128, eLN2b + l*128,
            mask, TK, cur, nullptr);
        merge_kernel<<<(size_t)BNKc*128/256,256,0,stream>>>(cur, inv, alt, hEbf);
        { float* t = cur; cur = alt; alt = t; }
        // -------- node update --------
        mlp3_kernel<1><<<BNKc/16/8,256,0,stream>>>(Vbf, hEbf, Eidx,
            base+oN1, nW1b + l*128, base+oN2, nW2b + l*128, base+oN3, nW3b + l*128,
            maskat, m3);
        node_reduce_ln_kernel<<<BNc,128,0,stream>>>(m3, Vf, nLN1s + l*128, nLN1b + l*128, Vf, Vbf);
        ffn_kernel<<<BNc/32,64,0,stream>>>(Vbf, base+oNFi, nFib + l*512,
            base+oNFo, nFob + l*128, m3);
        ln_kernel<<<BNc,128,0,stream>>>(Vf, m3, nLN2s + l*128, nLN2b + l*128,
            mask, 1, Vf, Vbf);
        // -------- token attention --------
        hin_kernel<<<(size_t)BNc*256/256,256,0,stream>>>(hT, Vf, hin);
        gemm128_kernel<<<BNc/128,256,0,stream>>>(hin, 256, base+oTK, nullptr, Kt);
        gemm128_kernel<<<BNc/128,256,0,stream>>>(hin, 256, base+oTV, nullptr, Vt);
        q_kernel<<<Bc,128,0,stream>>>(hT, tWQ + (size_t)l*128*128, Qb);
        attn_kernel<<<Bc*NHh,256,0,stream>>>(Qb, Kt, Vt, mask, updb);
        token_update_kernel<<<Bc,512,0,stream>>>(updb, hT,
            tWO + (size_t)l*128*128, tWOb + l*128,
            tLN1s + l*128, tLN1b + l*128,
            tFi + (size_t)l*128*512, tFib + l*512,
            tFo + (size_t)l*512*128, tFob + l*128,
            tLN2s + l*128, tLN2b + l*128, l, hT);
    }

    final_kernel<<<1,128,0,stream>>>(hT, Woutw, Woutb, (float*)d_out);
}
